// QuantumHyperNetwork_12704513262264
// MI455X (gfx1250) — matrix operand order repaired and checked by reading, not compiled, not run on hardware
//
#include <hip/hip_runtime.h>

// ---------------------------------------------------------------------------
// MI455X (gfx1250, wave32) implementation.
//
// Cost model:
//  - Quantum sim: 90 gates x 256KB of LDS traffic per block; the 128KB state
//    never leaves LDS (<= 320KB/WGP). HBM traffic = one 16MB probs write.
//  - GEMM1 (128x512 @ K=32768): mandatory HBM/L2 traffic is the 64MB W1 read
//    (L2-resident, 192MB L2) -> ~3us floor @ 23.3TB/s. f16 WMMA w/ fp32
//    accumulation keeps it memory-bound instead of fp32-matrix compute-bound.
//  - Split-K (disjoint K-slices -> no extra read traffic) lifts the GEMM grid
//    from 16 to 512 workgroups; partials combined with global_atomic_add_f32
//    into a zeroed fp32 accumulator. Double-buffered LDS -> 1 barrier per
//    64-deep K round, 2 WMMAs per round, global loads issued pre-barrier.
// ---------------------------------------------------------------------------

typedef __attribute__((ext_vector_type(16))) _Float16 v16h;
typedef __attribute__((ext_vector_type(8)))  _Float16 v8h;
typedef __attribute__((ext_vector_type(8)))  float    v8f;

#define NQ      15
#define QDIM    32768          // 2^15
#define BATCH   128
#define FEAT    256
#define NLAYERS 3

// =============================== quantum sim ===============================
// One block per batch element; full state vector lives in dynamic LDS.
__global__ __launch_bounds__(1024) void quantum_state_kernel(
    const float* __restrict__ feats,   // [BATCH, FEAT]
    const float* __restrict__ theta,   // [NLAYERS, NQ]
    const float* __restrict__ phi,     // [NLAYERS, NQ]
    float* __restrict__ probs)         // [BATCH, QDIM]
{
    extern __shared__ float st[];      // QDIM floats = 128 KB
    const int tid = threadIdx.x;
    const int b   = blockIdx.x;

    for (int i = tid; i < QDIM; i += 1024) st[i] = (i == 0) ? 1.0f : 0.0f;
    __syncthreads();

    for (int layer = 0; layer < NLAYERS; ++layer) {
        // ---- Ry(theta) on every qubit ----
        for (int q = 0; q < NQ; ++q) {
            const float ang = 0.5f * (theta[layer * NQ + q] +
                                      feats[b * FEAT + (q % FEAT)] * 0.1f);
            const float c = cosf(ang), s = sinf(ang);
            const int mlo = (1 << q) - 1;
            for (int p = tid; p < (QDIM >> 1); p += 1024) {
                const int i0 = ((p & ~mlo) << 1) | (p & mlo);   // bit q = 0
                const int i1 = i0 | (1 << q);                   // bit q = 1
                const float a0 = st[i0], a1 = st[i1];
                st[i0] = c * a0 - s * a1;
                st[i1] = s * a0 + c * a1;
            }
            __syncthreads();
        }
        // ---- circular CNOT chain ----
        for (int q = 0; q < NQ; ++q) {
            const int ctrl = q, tgt = (q + 1) % NQ;
            const int lo = ctrl < tgt ? ctrl : tgt;
            const int hi = ctrl < tgt ? tgt : ctrl;
            const int mlo = (1 << lo) - 1, mhi = (1 << hi) - 1;
            for (int p = tid; p < (QDIM >> 2); p += 1024) {
                int i = ((p & ~mlo) << 1) | (p & mlo);          // insert 0 @ lo
                i     = ((i & ~mhi) << 1) | (i & mhi);          // insert 0 @ hi
                const int i0 = i | (1 << ctrl);                 // ctrl=1, tgt=0
                const int i1 = i0 | (1 << tgt);                 // ctrl=1, tgt=1
                const float t0 = st[i0], t1 = st[i1];
                st[i0] = t1; st[i1] = t0;                       // flip target
            }
            __syncthreads();
        }
        // ---- Ry(phi) on every qubit ----
        for (int q = 0; q < NQ; ++q) {
            const float ang = 0.5f * (phi[layer * NQ + q] +
                                      feats[b * FEAT + ((q + 2) % FEAT)] * 0.1f);
            const float c = cosf(ang), s = sinf(ang);
            const int mlo = (1 << q) - 1;
            for (int p = tid; p < (QDIM >> 1); p += 1024) {
                const int i0 = ((p & ~mlo) << 1) | (p & mlo);
                const int i1 = i0 | (1 << q);
                const float a0 = st[i0], a1 = st[i1];
                st[i0] = c * a0 - s * a1;
                st[i1] = s * a0 + c * a1;
            }
            __syncthreads();
        }
    }

    // probs = state^2 (coalesced HBM write)
    for (int i = tid; i < QDIM; i += 1024) {
        const float v = st[i];
        probs[(size_t)b * QDIM + i] = v * v;
    }
}

// ======================== split-K WMMA f16 GEMM ============================
// acc[m,n] += sum_{k in slice} X[m,k] * W[n,k]      (fp32 global atomics)
// Block tile 64x64, 16 waves (4x4), each wave -> one 16x16 C tile.
// K advanced 64 per round: double-buffered LDS, ONE barrier per round,
// 2x v_wmma_f32_16x16x32_f16 per round, next round's global_load_b128s
// issued before the barrier to overlap HBM/L2 latency with compute.
#define MBLK  64
#define NBLK  64
#define KSTEP 64
#define LROW  80   // halfs per LDS row (160B stride: keeps b128 reads 16B-aligned)

__global__ __launch_bounds__(512) void gemm_f16_wmma_ksplit(
    const float* __restrict__ X,     // [M, K]
    const float* __restrict__ W,     // [N, K]
    float* __restrict__ acc,         // [M, N] fp32, pre-zeroed
    int M, int N, int K, int kslice) // kslice: multiple of 64
{
    __shared__ __align__(16) _Float16 Alds[2][MBLK * LROW];
    __shared__ __align__(16) _Float16 Blds[2][NBLK * LROW];

    const int tid  = threadIdx.x;
    const int lane = tid & 31;           // wave32
    const int wave = tid >> 5;           // 0..15
    const int wm   = wave >> 2;          // 0..3
    const int wn   = wave & 3;           // 0..3
    const int m0   = blockIdx.y * MBLK;
    const int n0   = blockIdx.x * NBLK;
    const int kbeg = blockIdx.z * kslice;
    const int kend = kbeg + kslice;

    // staging: 512 threads x 2 float4 per matrix = one 64x64 f32 tile
    const int srow = tid >> 3;           // 0..63
    const int scol = (tid & 7) * 4;      // 0,4,...,28

    const int half = lane >> 4;          // 0: lanes 0-15, 1: lanes 16-31
    const int lrow = lane & 15;

    const float* xrow = X + (size_t)(m0 + srow) * K + scol;
    const float* wrow = W + (size_t)(n0 + srow) * K + scol;

    // fragment bases (CDNA5 16-bit WMMA layouts)
    const int aoff = (wm * 16 + lrow) * LROW;
    const int boff = (wn * 16 + lrow) * LROW;
    const int akb  = half * 8;           // A: K{0..7,16..23} / K{8..15,24..31}
    const int bkb  = half * 8;           // B: K{0..7,16..23} / K{8..15,24..31}

    v8f accv = {};
    float4 ra0, ra1, rb0, rb1;

    auto load_regs = [&](int k) {
        ra0 = *(const float4*)(xrow + k);
        ra1 = *(const float4*)(xrow + k + 32);
        rb0 = *(const float4*)(wrow + k);
        rb1 = *(const float4*)(wrow + k + 32);
    };
    auto store_lds = [&](int buf) {
        _Float16* ap = &Alds[buf][srow * LROW + scol];
        ap[0]  = (_Float16)ra0.x; ap[1]  = (_Float16)ra0.y;
        ap[2]  = (_Float16)ra0.z; ap[3]  = (_Float16)ra0.w;
        ap[32] = (_Float16)ra1.x; ap[33] = (_Float16)ra1.y;
        ap[34] = (_Float16)ra1.z; ap[35] = (_Float16)ra1.w;
        _Float16* bp = &Blds[buf][srow * LROW + scol];
        bp[0]  = (_Float16)rb0.x; bp[1]  = (_Float16)rb0.y;
        bp[2]  = (_Float16)rb0.z; bp[3]  = (_Float16)rb0.w;
        bp[32] = (_Float16)rb1.x; bp[33] = (_Float16)rb1.y;
        bp[34] = (_Float16)rb1.z; bp[35] = (_Float16)rb1.w;
    };
    auto compute = [&](int buf) {
#pragma unroll
        for (int kk = 0; kk < KSTEP; kk += 32) {
            union { v16h v; v8h h[2]; } af, bf;
            const _Float16* ar = &Alds[buf][aoff + kk];
            af.h[0] = *(const v8h*)(ar + akb);
            af.h[1] = *(const v8h*)(ar + akb + 16);
            const _Float16* br = &Blds[buf][boff + kk];
            bf.h[0] = *(const v8h*)(br + bkb);
            bf.h[1] = *(const v8h*)(br + bkb + 16);
            accv = __builtin_amdgcn_wmma_f32_16x16x32_f16(
                false, af.v, false, bf.v, (short)0, accv, false, false);
        }
    };

    load_regs(kbeg);
    store_lds(0);
    int buf = 0;
    for (int k = kbeg + KSTEP; k < kend; k += KSTEP) {
        if (k + KSTEP < kend) {          // gfx1250 global_prefetch_b8
            __builtin_prefetch(wrow + k + KSTEP, 0, 1);
        }
        load_regs(k);                    // global loads in flight over barrier
        __syncthreads();                 // lds[buf] ready for everyone
        compute(buf);                    // WMMA from lds[buf]
        store_lds(buf ^ 1);              // fill other buffer
        buf ^= 1;
    }
    __syncthreads();
    compute(buf);

    // D layout: VGPR r -> row r (lanes 0-15) / r+8 (lanes 16-31), col = lane&15
    const int n     = n0 + wn * 16 + lrow;
    const int mrow0 = m0 + wm * 16 + half * 8;
#pragma unroll
    for (int r = 0; r < 8; ++r) {
        unsafeAtomicAdd(&acc[(size_t)(mrow0 + r) * N + n], accv[r]);
    }
}

// ======================== small elementwise kernels ========================
__global__ __launch_bounds__(256) void zero_kernel(float* __restrict__ p, int n) {
    const int i = blockIdx.x * 256 + threadIdx.x;
    if (i < n) p[i] = 0.0f;
}

__global__ __launch_bounds__(256) void bias_act_kernel(
    const float* __restrict__ acc, const float* __restrict__ bias,
    float* __restrict__ out, int total, int N, int do_relu) {
    const int i = blockIdx.x * 256 + threadIdx.x;
    if (i < total) {
        float v = acc[i] + bias[i & (N - 1)];   // N is a power of two here
        if (do_relu) v = fmaxf(v, 0.0f);
        out[i] = v;
    }
}

// ================================ launcher =================================
extern "C" void kernel_launch(void* const* d_in, const int* in_sizes, int n_in,
                              void* d_out, int out_size, void* d_ws, size_t ws_size,
                              hipStream_t stream) {
    const float* feats = (const float*)d_in[0];  // [128,256]
    const float* theta = (const float*)d_in[1];  // [3,15]
    const float* phi   = (const float*)d_in[2];  // [3,15]
    const float* W1    = (const float*)d_in[3];  // [512,32768]
    const float* b1    = (const float*)d_in[4];  // [512]
    const float* W2    = (const float*)d_in[5];  // [256,512]
    const float* b2    = (const float*)d_in[6];  // [256]
    const float* W3    = (const float*)d_in[7];  // [128,256]
    const float* b3    = (const float*)d_in[8];  // [128]
    float* out = (float*)d_out;                  // [128,128]

    // workspace layout (floats): probs | acc1 | acc2 | acc3 | H1 | H2
    float* probs = (float*)d_ws;                         // 128*32768
    float* acc1  = probs + (size_t)BATCH * QDIM;         // 128*512
    float* acc2  = acc1 + BATCH * 512;                   // 128*256
    float* acc3  = acc2 + BATCH * 256;                   // 128*128
    float* H1    = acc3 + BATCH * 128;                   // 128*512
    float* H2    = H1 + BATCH * 512;                     // 128*256

    // 0) zero the fp32 split-K accumulators (one contiguous region)
    const int accTotal = BATCH * (512 + 256 + 128);
    zero_kernel<<<(accTotal + 255) / 256, 256, 0, stream>>>(acc1, accTotal);

    // 1) quantum circuit: state in 128KB dynamic LDS, one block per batch elem
    quantum_state_kernel<<<BATCH, 1024, QDIM * sizeof(float), stream>>>(
        feats, theta, phi, probs);

    // 2) acc1 = probs @ W1^T   M=128 N=512 K=32768, split-K 32 -> 512 blocks
    gemm_f16_wmma_ksplit<<<dim3(512 / NBLK, BATCH / MBLK, 32), 512, 0, stream>>>(
        probs, W1, acc1, BATCH, 512, QDIM, QDIM / 32);
    bias_act_kernel<<<(BATCH * 512 + 255) / 256, 256, 0, stream>>>(
        acc1, b1, H1, BATCH * 512, 512, 1);

    // 3) acc2 = H1 @ W2^T      M=128 N=256 K=512, split-K 4
    gemm_f16_wmma_ksplit<<<dim3(256 / NBLK, BATCH / MBLK, 4), 512, 0, stream>>>(
        H1, W2, acc2, BATCH, 256, 512, 512 / 4);
    bias_act_kernel<<<(BATCH * 256 + 255) / 256, 256, 0, stream>>>(
        acc2, b2, H2, BATCH * 256, 256, 1);

    // 4) acc3 = H2 @ W3^T      M=128 N=128 K=256, split-K 2
    gemm_f16_wmma_ksplit<<<dim3(128 / NBLK, BATCH / MBLK, 2), 512, 0, stream>>>(
        H2, W3, acc3, BATCH, 128, 256, 256 / 2);
    bias_act_kernel<<<(BATCH * 128 + 255) / 256, 256, 0, stream>>>(
        acc3, b3, out, BATCH * 128, 128, 0);
}
